// Model_80848464379940
// MI455X (gfx1250) — matrix operand order repaired and checked by reading, not compiled, not run on hardware
//
#include <hip/hip_runtime.h>
#include <hip/hip_bf16.h>
#include <math.h>

typedef __bf16 bf16_t;
typedef bf16_t v16bf __attribute__((ext_vector_type(16)));
typedef bf16_t v8bf  __attribute__((ext_vector_type(8)));
typedef float  v8f   __attribute__((ext_vector_type(8)));
typedef int    v4i   __attribute__((vector_size(16)));   // matches builtin param type

#define NNODE 2048
#define BATCH 32
#define UDIM  64
#define LSEQ  12
#define HSEQ  12
#define NCPAD 2112            // padded B*C for diffusion GEMM (= 32*66)
#define KPAD  224             // padded 3*C rows for projection GEMM
#define XSROWS (BATCH*NNODE)  // 65536

#define BM 128
#define BN 64
#define LDA_S 40   // LDS row stride (elements) for A tile, padded
#define LDB_S 40   // LDS row stride (elements) for B tile (K-minor), padded

// ---- CDNA5 async global->LDS path (guarded; falls back to sync copy) ------
#if defined(__has_builtin)
#if __has_builtin(__builtin_amdgcn_global_load_async_to_lds_b128)
#define USE_ASYNC_LDS 1
#endif
#if __has_builtin(__builtin_amdgcn_s_wait_asynccnt)
#define HAVE_WAIT_ASYNC 1
#endif
#endif

__device__ __forceinline__ void cp16(bf16_t* ldst, const bf16_t* gsrc)
{
#if defined(USE_ASYNC_LDS)
    __builtin_amdgcn_global_load_async_to_lds_b128(
        (__attribute__((address_space(1))) v4i*)(gsrc),
        (__attribute__((address_space(3))) v4i*)(ldst), 0, 0);
#else
    *(uint4*)ldst = *(const uint4*)gsrc;
#endif
}

__device__ __forceinline__ void async_wait_all()
{
#if defined(USE_ASYNC_LDS)
#if defined(HAVE_WAIT_ASYNC)
    __builtin_amdgcn_s_wait_asynccnt(0);
#else
    asm volatile("s_wait_asynccnt 0x0" ::: "memory");
#endif
#endif
}

// ---------------------------------------------------------------------------
// Core WMMA GEMM: acc = A[M,K](bf16, row-major) @ Bt[N,K](bf16, K-minor).
// Double-buffered LDS staging via async global->LDS b128 copies.
// MODE 0: X1 = S@Xcat        -> X1t (K-minor, packed b128) + Xs block 1
// MODE 1: X2 = 2*S@X1 - Xcat -> Xs block 2 (Xcat read from Xct, packed)
// MODE 2: gate proj: sigmoid(acc+bg) -> rh = r*h (cols<64), u (cols>=64)
// MODE 3: cand proj: c = tanh(acc+bc); h = u*h + (1-u)*c
// ---------------------------------------------------------------------------
template<int MODE, int CC>
__global__ void __launch_bounds__(256)
wmma_gemm(const bf16_t* __restrict__ A, int lda,
          const bf16_t* __restrict__ Bt, int ldb, int K,
          bf16_t* __restrict__ Yt,
          bf16_t* __restrict__ Xs,
          const bf16_t* __restrict__ Xct,
          const float* __restrict__ bias,
          float* __restrict__ p0,
          float* __restrict__ p1,
          const float* __restrict__ hin)
{
    __shared__ __align__(16) bf16_t lA[2 * BM * LDA_S];
    __shared__ __align__(16) bf16_t lB[2 * BN * LDB_S];

    const int tid  = threadIdx.x;
    const int lane = tid & 31;
    const int wave = tid >> 5;
    const int wm   = wave & 3;           // 4 waves along M (32 rows each)
    const int wn   = wave >> 2;          // 2 waves along N (32 cols each)
    const int m0   = blockIdx.y * BM;
    const int n0   = blockIdx.x * BN;

    const int mrow = lane & 15;
    const int hi   = lane >> 4;          // 0: lanes 0-15, 1: lanes 16-31
    const int kA   = hi * 8;             // A frag K base (0 or 8)
    const int kB   = hi * 8;             // B frag K base (0 or 8)

    v8f acc[2][2] = {};

    auto stage = [&](int buf, int k0) {
        bf16_t* la = &lA[buf * (BM * LDA_S)];
        bf16_t* lb = &lB[buf * (BN * LDB_S)];
#pragma unroll
        for (int i = 0; i < 2; ++i) {                 // A: 512 x 16B chunks
            int chunk = tid + i * 256;
            int r = chunk >> 2, cc = (chunk & 3) << 3;
            cp16(&la[r * LDA_S + cc], A + (size_t)(m0 + r) * lda + (k0 + cc));
        }
        {                                             // B: 256 x 16B chunks
            int n = tid >> 2, cc = (tid & 3) << 3;
            cp16(&lb[n * LDB_S + cc], Bt + (size_t)(n0 + n) * ldb + (k0 + cc));
        }
    };

    const int iters = K >> 5;
    stage(0, 0);
    for (int it = 0; it < iters; ++it) {
        const int cur = it & 1;
        async_wait_all();
        __syncthreads();
        if (it + 1 < iters) stage(cur ^ 1, (it + 1) << 5);

        const bf16_t* la = &lA[cur * (BM * LDA_S)];
        const bf16_t* lb = &lB[cur * (BN * LDB_S)];

        v16bf afrag[2], bfrag[2];
#pragma unroll
        for (int tm = 0; tm < 2; ++tm) {
            int m = wm * 32 + tm * 16 + mrow;
            v8bf a0 = *(const v8bf*)(&la[m * LDA_S + kA]);       // K = kA..kA+7
            v8bf a1 = *(const v8bf*)(&la[m * LDA_S + kA + 16]);  // K = kA+16..kA+23
#pragma unroll
            for (int j = 0; j < 8; ++j) { afrag[tm][j] = a0[j]; afrag[tm][j + 8] = a1[j]; }
        }
#pragma unroll
        for (int tn = 0; tn < 2; ++tn) {
            int n = wn * 32 + tn * 16 + mrow;
            v8bf b0 = *(const v8bf*)(&lb[n * LDB_S + kB]);       // K = kB..kB+7
            v8bf b1 = *(const v8bf*)(&lb[n * LDB_S + kB + 16]);  // K = kB+16..kB+23
#pragma unroll
            for (int j = 0; j < 8; ++j) { bfrag[tn][j] = b0[j]; bfrag[tn][j + 8] = b1[j]; }
        }

#pragma unroll
        for (int tm = 0; tm < 2; ++tm)
#pragma unroll
            for (int tn = 0; tn < 2; ++tn)
                acc[tm][tn] = __builtin_amdgcn_wmma_f32_16x16x32_bf16(
                    false, afrag[tm], false, bfrag[tn],
                    (short)0, acc[tm][tn], false, false);
    }

    // ---- epilogue ----
#pragma unroll
    for (int tm = 0; tm < 2; ++tm) {
#pragma unroll
        for (int tn = 0; tn < 2; ++tn) {
            const int colg = n0 + wn * 32 + tn * 16 + mrow;
            const int rowb = m0 + wm * 32 + tm * 16 + hi * 8;    // 8 consecutive rows
            if (MODE == 0) {
                bf16_t pk[8];
#pragma unroll
                for (int j = 0; j < 8; ++j) pk[j] = (bf16_t)acc[tm][tn][j];
                *(uint4*)(Yt + (size_t)colg * NNODE + rowb) = *(uint4*)pk;   // X1t, packed
                const int b = colg / CC, c = colg - b * CC;
                if (b < BATCH) {
#pragma unroll
                    for (int j = 0; j < 8; ++j)
                        Xs[((size_t)(b * NNODE + rowb + j)) * KPAD + CC + c] = pk[j];
                }
            } else if (MODE == 1) {
                bf16_t xc8[8];
                *(uint4*)xc8 = *(const uint4*)(Xct + (size_t)colg * NNODE + rowb);
                const int b = colg / CC, c = colg - b * CC;
                if (b < BATCH) {
#pragma unroll
                    for (int j = 0; j < 8; ++j) {
                        float vv = 2.0f * acc[tm][tn][j] - (float)xc8[j];
                        Xs[((size_t)(b * NNODE + rowb + j)) * KPAD + 2 * CC + c] = (bf16_t)vv;
                    }
                }
            } else if (MODE == 2) {
#pragma unroll
                for (int j = 0; j < 8; ++j) {
                    float s = 1.0f / (1.0f + __expf(-(acc[tm][tn][j] + bias[colg])));
                    size_t base = (size_t)(rowb + j) * UDIM;
                    if (colg < UDIM) p0[base + colg] = s * hin[base + colg];   // rh = r*h
                    else             p1[base + (colg - UDIM)] = s;             // u
                }
            } else { // MODE == 3
#pragma unroll
                for (int j = 0; j < 8; ++j) {
                    float cnd = tanhf(acc[tm][tn][j] + bias[colg]);
                    size_t idx = (size_t)(rowb + j) * UDIM + colg;
                    float uu = p1[idx];
                    p0[idx] = uu * p0[idx] + (1.0f - uu) * cnd;                // h update
                }
            }
        }
    }
}

// ---------------------------------------------------------------------------
// Helper kernels
// ---------------------------------------------------------------------------
template<int CC, int CIN>
__global__ void pack_cat(const float* __restrict__ hist, int t,
                         const float* __restrict__ xin,
                         const float* __restrict__ hv,
                         bf16_t* __restrict__ Xcatt, bf16_t* __restrict__ Xs,
                         int total)
{
    int gid = blockIdx.x * blockDim.x + threadIdx.x;
    if (gid >= total) return;
    int n   = gid & (NNODE - 1);
    int col = gid >> 11;
    int b   = col / CC;
    int c   = col - b * CC;
    float v;
    if (c < CIN) {
        v = (CIN == 2) ? hist[(((size_t)b * LSEQ + t) * NNODE + n) * 2 + c]
                       : xin[(size_t)b * NNODE + n];
    } else {
        v = hv[((size_t)b * NNODE + n) * UDIM + (c - CIN)];
    }
    bf16_t bv = (bf16_t)v;
    Xcatt[(size_t)col * NNODE + n] = bv;               // K-minor layout
    Xs[((size_t)(b * NNODE + n)) * KPAD + c] = bv;
}

__global__ void pad_cols_bf16(bf16_t* __restrict__ buf, int ld, int c0, int c1, int rows)
{
    int gid = blockIdx.x * blockDim.x + threadIdx.x;
    int w = c1 - c0;
    if (gid >= rows * w) return;
    int r = gid / w;
    int c = c0 + (gid - r * w);
    buf[(size_t)r * ld + c] = (bf16_t)0.0f;
}

__global__ void zero_bf16(bf16_t* __restrict__ p, int n)
{
    int gid = blockIdx.x * blockDim.x + threadIdx.x;
    if (gid < n) p[gid] = (bf16_t)0.0f;
}

__global__ void prep_w(const float* __restrict__ W, bf16_t* __restrict__ Wrt, int C, int NO)
{
    int gid = blockIdx.x * blockDim.x + threadIdx.x;   // gid = j*KPAD + kr
    if (gid >= KPAD * NO) return;
    int j  = gid / KPAD;
    int kr = gid - j * KPAD;
    float v = 0.0f;
    if (kr < 3 * C) {
        int k = kr / C, c = kr - k * C;
        v = W[(size_t)(c * 3 + k) * NO + j];           // reference row order = c*3+k
    }
    Wrt[gid] = (bf16_t)v;                              // Wrt[NO][KPAD], K-minor
}

__global__ void f32_to_bf16(const float* __restrict__ src, bf16_t* __restrict__ dst, int n)
{
    int gid = blockIdx.x * blockDim.x + threadIdx.x;
    if (gid < n) dst[gid] = (bf16_t)src[gid];
}

__global__ void zero_f32(float* __restrict__ p, int n)
{
    int gid = blockIdx.x * blockDim.x + threadIdx.x;
    if (gid < n) p[gid] = 0.0f;
}

__global__ void proj_out(const float* __restrict__ h, const float* __restrict__ Wp,
                         const float* __restrict__ bp, float* __restrict__ out,
                         float* __restrict__ xin, int t)
{
    int row = blockIdx.x * blockDim.x + threadIdx.x;   // 65536 rows = (b,n)
    if (row >= XSROWS) return;
    float acc = bp[0];
    const float4* h4 = (const float4*)(h + (size_t)row * UDIM);
    const float4* w4 = (const float4*)Wp;
#pragma unroll
    for (int i = 0; i < UDIM / 4; ++i) {
        float4 a = h4[i], w = w4[i];
        acc += a.x * w.x + a.y * w.y + a.z * w.z + a.w * w.w;
    }
    int b = row >> 11, n = row & (NNODE - 1);
    out[((size_t)b * HSEQ + t) * NNODE + n] = acc;     // [B,H,N,1]
    xin[row] = acc;                                    // decoder feedback
}

// ---------------------------------------------------------------------------
extern "C" void kernel_launch(void* const* d_in, const int* in_sizes, int n_in,
                              void* d_out, int out_size, void* d_ws, size_t ws_size,
                              hipStream_t stream)
{
    const float* hist = (const float*)d_in[0];
    const float* supp = (const float*)d_in[1];
    const float* eWg  = (const float*)d_in[2];
    const float* ebg  = (const float*)d_in[3];
    const float* eWc  = (const float*)d_in[4];
    const float* ebc  = (const float*)d_in[5];
    const float* dWg  = (const float*)d_in[6];
    const float* dbg  = (const float*)d_in[7];
    const float* dWc  = (const float*)d_in[8];
    const float* dbc  = (const float*)d_in[9];
    const float* Wp   = (const float*)d_in[10];
    const float* bp   = (const float*)d_in[11];
    float* out = (float*)d_out;

    char* ws = (char*)d_ws;
    size_t off = 0;
    auto carve = [&](size_t bytes) -> char* {
        char* p = ws + off;
        off += (bytes + 255) & ~(size_t)255;
        return p;
    };
    bf16_t* Sbf   = (bf16_t*)carve((size_t)NNODE * NNODE * 2);
    bf16_t* Xct   = (bf16_t*)carve((size_t)NCPAD * NNODE * 2);   // Xcat, K-minor
    bf16_t* X1t   = (bf16_t*)carve((size_t)NCPAD * NNODE * 2);   // X1,  K-minor
    bf16_t* Xsb   = (bf16_t*)carve((size_t)XSROWS * KPAD * 2);
    bf16_t* Wr_eg = (bf16_t*)carve((size_t)128 * KPAD * 2);
    bf16_t* Wr_ec = (bf16_t*)carve((size_t)64  * KPAD * 2);
    bf16_t* Wr_dg = (bf16_t*)carve((size_t)128 * KPAD * 2);
    bf16_t* Wr_dc = (bf16_t*)carve((size_t)64  * KPAD * 2);
    float*  hbuf  = (float*)carve((size_t)XSROWS * UDIM * 4);
    float*  ubuf  = (float*)carve((size_t)XSROWS * UDIM * 4);
    float*  rhbuf = (float*)carve((size_t)XSROWS * UDIM * 4);
    float*  xin   = (float*)carve((size_t)XSROWS * 4);

    dim3 blk(256);
    f32_to_bf16<<<(NNODE * NNODE + 255) / 256, blk, 0, stream>>>(supp, Sbf, NNODE * NNODE);
    prep_w<<<(KPAD * 128 + 255) / 256, blk, 0, stream>>>(eWg, Wr_eg, 66, 128);
    prep_w<<<(KPAD * 64  + 255) / 256, blk, 0, stream>>>(eWc, Wr_ec, 66, 64);
    prep_w<<<(KPAD * 128 + 255) / 256, blk, 0, stream>>>(dWg, Wr_dg, 65, 128);
    prep_w<<<(KPAD * 64  + 255) / 256, blk, 0, stream>>>(dWc, Wr_dc, 65, 64);
    zero_f32<<<(XSROWS * UDIM + 255) / 256, blk, 0, stream>>>(hbuf, XSROWS * UDIM);
    zero_f32<<<(XSROWS + 255) / 256, blk, 0, stream>>>(xin, XSROWS);

    dim3 gD(NCPAD / BN, NNODE / BM);     // 33 x 16  (diffusion GEMMs)
    dim3 gPg(128 / BN, XSROWS / BM);     // 2 x 512  (gate projection)
    dim3 gPc(64 / BN, XSROWS / BM);      // 1 x 512  (candidate projection)

    for (int phase = 0; phase < 2; ++phase) {
        const int steps = (phase == 0) ? LSEQ : HSEQ;
        const int C     = (phase == 0) ? 66 : 65;
        const bf16_t* Wg = (phase == 0) ? Wr_eg : Wr_dg;
        const bf16_t* Wc = (phase == 0) ? Wr_ec : Wr_dc;
        const float* bg  = (phase == 0) ? ebg : dbg;
        const float* bc  = (phase == 0) ? ebc : dbc;
        const int totalPack = NNODE * BATCH * C;

        for (int t = 0; t < steps; ++t) {
            for (int g = 0; g < 2; ++g) {   // g=0: gate (h), g=1: candidate (r*h)
                const float* hv = (g == 0) ? hbuf : rhbuf;
                if (phase == 0)
                    pack_cat<66, 2><<<(totalPack + 255) / 256, blk, 0, stream>>>(
                        hist, t, xin, hv, Xct, Xsb, totalPack);
                else {
                    pack_cat<65, 1><<<(totalPack + 255) / 256, blk, 0, stream>>>(
                        hist, t, xin, hv, Xct, Xsb, totalPack);
                    // zero Xcat K-minor rows [B*C, NCPAD): contiguous tail region
                    zero_bf16<<<((NCPAD - BATCH * 65) * NNODE + 255) / 256, blk, 0, stream>>>(
                        Xct + (size_t)(BATCH * 65) * NNODE, (NCPAD - BATCH * 65) * NNODE);
                }
                // zero Xs pad cols [3C, KPAD)
                pad_cols_bf16<<<(XSROWS * (KPAD - 3 * C) + 255) / 256, blk, 0, stream>>>(
                    Xsb, KPAD, 3 * C, KPAD, XSROWS);

                // X1 = S @ Xcat  (store X1t + Xs block 1)
                if (phase == 0)
                    wmma_gemm<0, 66><<<gD, blk, 0, stream>>>(
                        Sbf, NNODE, Xct, NNODE, NNODE,
                        X1t, Xsb, nullptr, nullptr, nullptr, nullptr, nullptr);
                else
                    wmma_gemm<0, 65><<<gD, blk, 0, stream>>>(
                        Sbf, NNODE, Xct, NNODE, NNODE,
                        X1t, Xsb, nullptr, nullptr, nullptr, nullptr, nullptr);
                // X2 = 2*S @ X1 - Xcat  (Xs block 2)
                if (phase == 0)
                    wmma_gemm<1, 66><<<gD, blk, 0, stream>>>(
                        Sbf, NNODE, X1t, NNODE, NNODE,
                        nullptr, Xsb, Xct, nullptr, nullptr, nullptr, nullptr);
                else
                    wmma_gemm<1, 65><<<gD, blk, 0, stream>>>(
                        Sbf, NNODE, X1t, NNODE, NNODE,
                        nullptr, Xsb, Xct, nullptr, nullptr, nullptr, nullptr);

                if (g == 0) {
                    // ru = sigmoid(Xs @ Wg + bg): rh, u
                    wmma_gemm<2, 1><<<gPg, blk, 0, stream>>>(
                        Xsb, KPAD, Wg, KPAD, KPAD,
                        nullptr, nullptr, nullptr, bg, rhbuf, ubuf, hbuf);
                } else {
                    // c = tanh(Xs @ Wc + bc); h = u*h + (1-u)*c
                    wmma_gemm<3, 1><<<gPc, blk, 0, stream>>>(
                        Xsb, KPAD, Wc, KPAD, KPAD,
                        nullptr, nullptr, nullptr, bc, hbuf, ubuf, nullptr);
                }
            }
            if (phase == 1)
                proj_out<<<(XSROWS + 255) / 256, blk, 0, stream>>>(hbuf, Wp, bp, out, xin, t);
        }
    }
}
